// GHMCCosineLoss_61718680043545
// MI455X (gfx1250) — matrix operand order repaired and checked by reading, not compiled, not run on hardware
//
#include <hip/hip_runtime.h>
#include <hip/hip_bf16.h>

typedef __attribute__((ext_vector_type(4)))  _Float16 v4h;
typedef __attribute__((ext_vector_type(8)))  _Float16 v8h;
typedef __attribute__((ext_vector_type(16))) _Float16 v16h;
typedef __attribute__((ext_vector_type(8)))  float    v8f;

#define NROWS 4096
#define NDIM  512
#define NBINS 10
#define BPAD  21   // row stride for per-lane LDS bins; gcd(21,64)=1 -> no bank aliasing

// ---------------------------------------------------------------------------
// d_ws layout:
//   [0 .. 20 floats)          : global bins  (cnt[10], s2[10])
//   [256 bytes ...)           : xn (f16, 4096 x 512, row-major) = 4 MB
// ---------------------------------------------------------------------------

__global__ void ghm_init_bins(float* __restrict__ bins) {
    int t = threadIdx.x;
    if (t < 2 * NBINS) bins[t] = 0.0f;
}

// One block (128 threads) per row: fp32 L2 norm, scale, convert to f16.
__global__ void ghm_normalize_rows(const float* __restrict__ x,
                                   _Float16* __restrict__ xn) {
    const int row = blockIdx.x;
    const int tid = threadIdx.x;  // 0..127
    const float4 v = ((const float4*)(x + (size_t)row * NDIM))[tid];
    float s = v.x * v.x + v.y * v.y + v.z * v.z + v.w * v.w;

    __shared__ float red[128];
    red[tid] = s;
    __syncthreads();
    #pragma unroll
    for (int off = 64; off > 0; off >>= 1) {
        if (tid < off) red[tid] += red[tid + off];
        __syncthreads();
    }
    const float inv = 1.0f / fmaxf(sqrtf(red[0]), 1e-12f);

    v4h h;
    h[0] = (_Float16)(v.x * inv);
    h[1] = (_Float16)(v.y * inv);
    h[2] = (_Float16)(v.z * inv);
    h[3] = (_Float16)(v.w * inv);
    *(v4h*)(xn + (size_t)row * NDIM + tid * 4) = h;
}

// Block tile 128x128, 8 waves (wave32) as 4(M) x 2(N); each wave: 32x64 via
// 2x4 WMMA 16x16 f32 accumulators. Fused branchless histogram epilogue into
// per-lane LDS bins (no intra-wave same-address atomic serialization).
__global__ __launch_bounds__(256)
void ghm_gram_hist(const _Float16* __restrict__ xn,
                   const int* __restrict__ tg,
                   float* __restrict__ bins) {
    const int bx = blockIdx.x, by = blockIdx.y;
    if (bx < by) return;  // symmetric Gram: keep block-upper triangle only

    const int tid  = threadIdx.x;
    const int lane = tid & 31;
    const int wave = tid >> 5;      // 0..7
    const int wM   = wave & 3;      // 4 waves along M
    const int wN   = wave >> 2;     // 2 waves along N
    const int lmod = lane & 15;
    const int lhi  = lane >> 4;     // 0 or 1

    const int rowBase = by * 128 + wM * 32;  // wave covers rows [rowBase, +32)
    const int colBase = bx * 128 + wN * 64;  // wave covers cols [colBase, +64)

    // Per-lane private bin rows: lane l of every wave uses row l. Intra-wave
    // ds_add addresses are unique (and bank-conflict-free with stride 21);
    // only occasional cross-wave same-lane collisions remain.
    __shared__ float s_cnt[32][BPAD];
    __shared__ float s_s2[32][BPAD];
    for (int i = tid; i < 32 * BPAD; i += 256) {
        (&s_cnt[0][0])[i] = 0.0f;
        (&s_s2[0][0])[i]  = 0.0f;
    }
    __syncthreads();

    // Fragment base pointers (per lane).
    // A (contiguous k order, same as B): lane holds row (base+lmod), halves 0..7 at
    // k = lhi*16, halves 8..15 at k = lhi*16 + 8.
    const _Float16* aRow[2];
    aRow[0] = xn + (size_t)(rowBase + lmod) * NDIM + lhi * 16;
    aRow[1] = aRow[0] + (size_t)16 * NDIM;
    // B (contiguous k order, same as A): lane holds column (base+lmod) of B, i.e. row of
    // xn, 16 contiguous K values starting at k = lhi*16.
    const _Float16* bCol[4];
    {
        const _Float16* b0 = xn + (size_t)(colBase + lmod) * NDIM + lhi * 16;
        bCol[0] = b0;
        bCol[1] = b0 + (size_t)16 * NDIM;
        bCol[2] = b0 + (size_t)32 * NDIM;
        bCol[3] = b0 + (size_t)48 * NDIM;
    }

    v8f acc[2][4];
    const v8f vzero = {0.f, 0.f, 0.f, 0.f, 0.f, 0.f, 0.f, 0.f};
    #pragma unroll
    for (int mi = 0; mi < 2; ++mi)
        #pragma unroll
        for (int ni = 0; ni < 4; ++ni)
            acc[mi][ni] = vzero;

    for (int k0 = 0; k0 < NDIM; k0 += 32) {
        v16h a[2], b[4];
        #pragma unroll
        for (int mi = 0; mi < 2; ++mi) {
            v8h lo = *(const v8h*)(aRow[mi] + k0);
            v8h hi = *(const v8h*)(aRow[mi] + k0 + 8);
            #pragma unroll
            for (int i = 0; i < 8; ++i) { a[mi][i] = lo[i]; a[mi][i + 8] = hi[i]; }
        }
        #pragma unroll
        for (int ni = 0; ni < 4; ++ni)
            b[ni] = *(const v16h*)(bCol[ni] + k0);

        #pragma unroll
        for (int mi = 0; mi < 2; ++mi)
            #pragma unroll
            for (int ni = 0; ni < 4; ++ni)
                acc[mi][ni] = __builtin_amdgcn_wmma_f32_16x16x32_f16(
                    /*neg_a=*/false, a[mi], /*neg_b=*/false, b[ni],
                    /*c_mod=*/(short)0, acc[mi][ni],
                    /*reuse_a=*/false, /*reuse_b=*/false);
    }

    // Preload targets covering this wave's tile.
    // C layout: VGPR v, lane l -> row = v + 8*(l/16), col = l%16 (per 16x16).
    int trow[2][8];
    #pragma unroll
    for (int mi = 0; mi < 2; ++mi)
        #pragma unroll
        for (int v = 0; v < 8; ++v)
            trow[mi][v] = tg[rowBase + mi * 16 + lhi * 8 + v];
    int tcol[4];
    #pragma unroll
    for (int ni = 0; ni < 4; ++ni)
        tcol[ni] = tg[colBase + ni * 16 + lmod];

    const float scale = (bx == by) ? 1.0f : 2.0f;  // off-diag blocks count twice
    const float edge_last = 1.0f + 1e-6f;
    float* const myCnt = &s_cnt[lane][0];
    float* const myS2  = &s_s2[lane][0];

    #pragma unroll
    for (int mi = 0; mi < 2; ++mi) {
        #pragma unroll
        for (int ni = 0; ni < 4; ++ni) {
            #pragma unroll
            for (int v = 0; v < 8; ++v) {
                float c     = acc[mi][ni][v];
                float label = (trow[mi][v] == tcol[ni]) ? 1.0f : 0.0f;
                float d     = c - label;
                float g     = fabsf(d);
                // Branchless: clamp idx into [0,9] (safe for any g in [0,2]),
                // zero the contribution when g falls past the last edge.
                int   idx     = (int)(g * 10.0f);
                idx           = idx > 9 ? 9 : idx;
                float contrib = (g < edge_last) ? scale : 0.0f;
                atomicAdd(&myCnt[idx], contrib);
                atomicAdd(&myS2[idx],  contrib * d * d);
            }
        }
    }
    __syncthreads();

    // Column-sum the 32 lane-private rows, one thread per bin, then flush.
    if (tid < NBINS) {
        float c0 = 0.0f, s0 = 0.0f;
        #pragma unroll
        for (int l = 0; l < 32; ++l) {
            c0 += s_cnt[l][tid];
            s0 += s_s2[l][tid];
        }
        atomicAdd(&bins[tid],         c0);
        atomicAdd(&bins[NBINS + tid], s0);
    }
}

__global__ void ghm_finalize(const float* __restrict__ bins,
                             const float* __restrict__ acc_sum,
                             float* __restrict__ out) {
    if (threadIdx.x == 0 && blockIdx.x == 0) {
        const float tot = 16777216.0f;  // B*B = 4096^2
        float valid_num = 0.0f;
        float loss = 0.0f;
        #pragma unroll
        for (int b = 0; b < NBINS; ++b) {
            float cnt  = bins[b];
            float accn = 0.5f * acc_sum[b] + 0.5f * cnt;  // momentum = 0.5
            float w    = tot / (accn + 1e-6f);
            loss += bins[NBINS + b] * w;
            valid_num += cnt;
        }
        if (valid_num > 0.0f) loss /= valid_num;
        out[0] = loss / tot;  // mean over B*B elements
    }
}

extern "C" void kernel_launch(void* const* d_in, const int* in_sizes, int n_in,
                              void* d_out, int out_size, void* d_ws, size_t ws_size,
                              hipStream_t stream) {
    const float* x       = (const float*)d_in[0];  // [4096, 512] f32
    const float* acc_sum = (const float*)d_in[1];  // [10] f32
    const int*   targets = (const int*)d_in[2];    // [4096] int
    float*       out     = (float*)d_out;          // [1] f32

    float*    bins = (float*)d_ws;                       // 20 floats
    _Float16* xn   = (_Float16*)((char*)d_ws + 256);     // 4096*512 f16

    ghm_init_bins<<<1, 32, 0, stream>>>(bins);
    ghm_normalize_rows<<<NROWS, 128, 0, stream>>>(x, xn);
    dim3 grid(NROWS / 128, NROWS / 128);  // 32 x 32 blocks (lower half early-exits)
    ghm_gram_hist<<<grid, 256, 0, stream>>>(xn, targets, bins);
    ghm_finalize<<<1, 32, 0, stream>>>(bins, acc_sum, out);
}
